// CustomSRUCell_11605001633987
// MI455X (gfx1250) — matrix operand order repaired and checked by reading, not compiled, not run on hardware
//
#include <hip/hip_runtime.h>
#include <math.h>

// MI455X / gfx1250: wave32, WMMA 16x16x32 bf16 (fp32 accumulate), 320KB LDS/WGP,
// async global->LDS DMA (ASYNCcnt), Tensor Data Mover (TENSORcnt).

typedef __attribute__((ext_vector_type(16))) __bf16 v16bf;
typedef __attribute__((ext_vector_type(8)))  float  v8f;

#define B_DIM   64
#define T_LEN   1024
#define D_DIM   512
#define U_DIM   512
#define M_TOTAL (B_DIM * T_LEN)   // 65536
#define K3U     (3 * U_DIM)       // 1536

// fp32 -> bf16, round-to-nearest-even
static __device__ __forceinline__ unsigned short f2bf(float f) {
    unsigned int u = __float_as_uint(f);
    unsigned int r = (u + 0x7FFFu + ((u >> 16) & 1u)) >> 16;
    return (unsigned short)r;
}

static __device__ __forceinline__ v16bf mk_frag(uint4 lo, uint4 hi) {
    union { uint4 u[2]; v16bf v; } x;
    x.u[0] = lo; x.u[1] = hi;
    return x.v;
}

// v_exp_f32 + v_rcp_f32 fast paths (TRANS ops co-execute with WMMA)
static __device__ __forceinline__ float fast_sigmoid(float x) {
    return __builtin_amdgcn_rcpf(1.0f + __expf(-x));
}
static __device__ __forceinline__ float fast_tanh(float x) {
    float e = __expf(2.0f * x);                 // +inf / 0 saturate correctly
    return 1.0f - 2.0f * __builtin_amdgcn_rcpf(e + 1.0f);
}

// CDNA5 async DMA: global -> LDS, 16B per lane, tracked by ASYNCcnt.
static __device__ __forceinline__ void async_b128(unsigned lds_addr, unsigned byte_off,
                                                  const void* base) {
    asm volatile("global_load_async_to_lds_b128 %0, %1, %2"
                 :: "v"(lds_addr), "v"(byte_off), "s"(base) : "memory");
}
static __device__ __forceinline__ void wait_async0() {
    asm volatile("s_wait_asynccnt 0" ::: "memory");
}

// ---------------------------------------------------------------------------
// Phase 0a: fp32 -> bf16 bulk convert (8 elements / thread)
// ---------------------------------------------------------------------------
__global__ void cvt_bf16_kernel(const float* __restrict__ src,
                                unsigned short* __restrict__ dst, int n8) {
    int i = blockIdx.x * blockDim.x + threadIdx.x;
    if (i >= n8) return;
    const float4* s = (const float4*)src + (size_t)i * 2;
    float4 a = s[0], b = s[1];
    uint4 o;
    o.x = (unsigned)f2bf(a.x) | ((unsigned)f2bf(a.y) << 16);
    o.y = (unsigned)f2bf(a.z) | ((unsigned)f2bf(a.w) << 16);
    o.z = (unsigned)f2bf(b.x) | ((unsigned)f2bf(b.y) << 16);
    o.w = (unsigned)f2bf(b.z) | ((unsigned)f2bf(b.w) << 16);
    ((uint4*)dst)[i] = o;
}

// ---------------------------------------------------------------------------
// Phase 0b: transpose + convert: src[rows][cols] fp32 -> dst[cols][rows] bf16
// ---------------------------------------------------------------------------
__global__ void tconv_kernel(const float* __restrict__ src,
                             unsigned short* __restrict__ dst,
                             int rows, int cols) {
    int i = blockIdx.x * blockDim.x + threadIdx.x;
    if (i >= rows * cols) return;
    int c = i / rows;
    int r = i - c * rows;
    dst[(size_t)c * rows + r] = f2bf(src[(size_t)r * cols + c]);
}

// ---------------------------------------------------------------------------
// Phase 1: gates = act(x @ W_gate + bias)   [65536,512] @ [512,1536]
// 128x128x32 tiles, double-buffered async global->LDS, 1 barrier / K-step.
// ---------------------------------------------------------------------------
__launch_bounds__(256)
__global__ void gates_gemm_kernel(const unsigned short* __restrict__ xb,   // [M,512] bf16
                                  const unsigned short* __restrict__ wgT,  // [1536,512] bf16 N-major
                                  const float* __restrict__ bias,
                                  float* __restrict__ fo,
                                  float* __restrict__ uo,
                                  float* __restrict__ co) {
    __shared__ __align__(16) unsigned short As[2][128 * 32];
    __shared__ __align__(16) unsigned short Bs[2][128 * 32];

    const int tid = threadIdx.x;
    const int wid = tid >> 5, lane = tid & 31;
    const int wm = wid >> 2, wn = wid & 3;       // wave grid 2 (M) x 4 (N)
    const int row0 = blockIdx.y * 128;
    const int col0 = blockIdx.x * 128;
    const int hs = (lane < 16) ? 0 : 1;
    const int lr = lane & 15;

    const int v0 = tid, v1 = tid + 256;
    const int r0 = v0 >> 2, kc0 = (v0 & 3) << 3;
    const int r1 = v1 >> 2, kc1 = (v1 & 3) << 3;

    auto issue_tile = [&](int kb, int buf) {
        unsigned a0 = (unsigned)(uintptr_t)&As[buf][r0 * 32 + kc0];
        unsigned a1 = (unsigned)(uintptr_t)&As[buf][r1 * 32 + kc1];
        unsigned b0 = (unsigned)(uintptr_t)&Bs[buf][r0 * 32 + kc0];
        unsigned b1 = (unsigned)(uintptr_t)&Bs[buf][r1 * 32 + kc1];
        async_b128(a0, (unsigned)(((size_t)(row0 + r0) * 512 + kb * 32 + kc0) * 2), xb);
        async_b128(a1, (unsigned)(((size_t)(row0 + r1) * 512 + kb * 32 + kc1) * 2), xb);
        async_b128(b0, (unsigned)(((size_t)(col0 + r0) * 512 + kb * 32 + kc0) * 2), wgT);
        async_b128(b1, (unsigned)(((size_t)(col0 + r1) * 512 + kb * 32 + kc1) * 2), wgT);
    };

    v8f acc[4][2];
#pragma unroll
    for (int mi = 0; mi < 4; ++mi)
#pragma unroll
        for (int ni = 0; ni < 2; ++ni) acc[mi][ni] = {};

    issue_tile(0, 0);
    for (int kb = 0; kb < 16; ++kb) {
        const int buf = kb & 1;
        wait_async0();          // current buffer's DMA done (this wave)
        __syncthreads();        // ... and everyone else's
        if (kb + 1 < 16) issue_tile(kb + 1, buf ^ 1);  // overlaps WMMA below

        v16bf bfrag[2];
#pragma unroll
        for (int ni = 0; ni < 2; ++ni) {
            const unsigned short* bp = &Bs[buf][(wn * 32 + ni * 16 + lr) * 32 + hs * 16];
            bfrag[ni] = mk_frag(*(const uint4*)bp, *(const uint4*)(bp + 8));
        }
#pragma unroll
        for (int mi = 0; mi < 4; ++mi) {
            const unsigned short* ap = &As[buf][(wm * 64 + mi * 16 + lr) * 32 + hs * 16];
            v16bf afrag = mk_frag(*(const uint4*)ap, *(const uint4*)(ap + 8));
#pragma unroll
            for (int ni = 0; ni < 2; ++ni)
                acc[mi][ni] = __builtin_amdgcn_wmma_f32_16x16x32_bf16(
                    false, afrag, false, bfrag[ni], (short)0, acc[mi][ni], false, false);
        }
    }

    const int gate = col0 >> 9;  // 0 = f (sigmoid), 1 = u (sigmoid), 2 = cand (gelu)
    float* outp = (gate == 0) ? fo : (gate == 1) ? uo : co;
#pragma unroll
    for (int mi = 0; mi < 4; ++mi) {
#pragma unroll
        for (int ni = 0; ni < 2; ++ni) {
            int n = col0 + wn * 32 + ni * 16 + lr;
            float bv = bias[n];
            int ncol = n & 511;
#pragma unroll
            for (int j = 0; j < 8; ++j) {
                int m = row0 + wm * 64 + mi * 16 + j + hs * 8;
                float g = acc[mi][ni][j] + bv;
                float r;
                if (gate == 2) r = 0.5f * g * (1.0f + erff(g * 0.70710678118654752f));
                else           r = fast_sigmoid(g);
                outp[(size_t)m * 512 + ncol] = r;
            }
        }
    }
}

// ---------------------------------------------------------------------------
// Phase 2: scan. Batch rows are INDEPENDENT -> 8 workgroups x 8 rows, no
// inter-WG sync. Per WG LDS: Wm K-half resident (256KB bf16, TDM-DMA'd),
// C fp32 (16KB) + m fp32 (16KB) + C bf16 padded to 16 rows (16KB) = 304KB.
// 256 threads = 8 waves; per step: LN -> [8(16),512]@[512,512] WMMA -> LN.
// ---------------------------------------------------------------------------
__launch_bounds__(256)
__global__ void sru_scan_kernel(const float* __restrict__ fg,
                                const float* __restrict__ ug,
                                const float* __restrict__ cg,
                                const unsigned short* __restrict__ wmT,  // [512,512] bf16 N-major
                                const float* __restrict__ gamma_c, const float* __restrict__ beta_c,
                                const float* __restrict__ gamma_m, const float* __restrict__ beta_m,
                                float* __restrict__ out) {
    extern __shared__ __align__(16) char smem[];
    unsigned short* WmL = (unsigned short*)smem;                            // [512][256] 256KB
    float* Cs = (float*)(smem + (size_t)512 * 256 * 2);                     // [8][512]  16KB
    float* Ms = (float*)(smem + (size_t)512 * 256 * 2 + 8 * 512 * 4);       // [8][512]  16KB
    unsigned short* Cb =
        (unsigned short*)(smem + (size_t)512 * 256 * 2 + 2 * 8 * 512 * 4);  // [16][512] 16KB

    const int tid = threadIdx.x;
    const int w = tid >> 5, lane = tid & 31;
    const int hs = (lane < 16) ? 0 : 1;
    const int lr = lane & 15;
    const int bbase = blockIdx.x * 8;

    // ---- Stage resident K-half of Wm (k in [0,256)) into LDS ----
    // Tensor Data Mover: one 2D tile DMA (tile 256 x 512 rows of a 512-stride
    // tensor, 2B elements), issued by wave 0, tracked by TENSORcnt.
    if (w == 0) {
        typedef __attribute__((ext_vector_type(4))) unsigned tdm_v4u;
        typedef __attribute__((ext_vector_type(8))) int      tdm_v8i;
        typedef __attribute__((ext_vector_type(4))) int      tdm_v4i;
        unsigned long long ga = (unsigned long long)(uintptr_t)wmT;
        tdm_v4u g0;
        g0[0] = 1u;                                          // count=1, user D#
        g0[1] = (unsigned)(uintptr_t)WmL;                    // lds_addr (bytes)
        g0[2] = (unsigned)(ga & 0xFFFFFFFFu);                // global_addr[31:0]
        g0[3] = (unsigned)((ga >> 32) & 0x01FFFFFFu)         // global_addr[56:32]
              | (2u << 30);                                  // type = 2 ("image")
        tdm_v8i g1;
        g1[0] = (int)(1u << 16);          // workgroup_mask=0, data_size=1 (2B)
        g1[1] = (int)(512u << 16);        // tensor_dim0 = 512 (bits 79:48, lo16)
        g1[2] = (int)(512u << 16);        // dim0 hi16=0 | tensor_dim1 = 512 (lo16)
        g1[3] = (int)(256u << 16);        // dim1 hi16=0 | tile_dim0 = 256
        g1[4] = (int)512u;                // tile_dim1 = 512, tile_dim2 = 0
        g1[5] = (int)512u;                // tensor_dim0_stride = 512 (lo32)
        g1[6] = 0;                        // stride0 hi16 = 0, stride1 lo16 = 0
        g1[7] = 0;
        tdm_v4i gz4 = {0, 0, 0, 0};       // 2D tensor: groups 2/3 unused
        tdm_v8i gz8 = {0, 0, 0, 0, 0, 0, 0, 0};
        // 6-arg toolchain form: (g0, g1, g2, g3, extra, cpol)
        __builtin_amdgcn_tensor_load_to_lds(g0, g1, gz4, gz4, gz8, 0);
        __builtin_amdgcn_s_wait_tensorcnt(0);
    }
    for (int i = tid; i < 16 * 512; i += 256) Cb[i] = 0;   // incl. pad rows 8..15
    for (int i = tid; i < 8 * 512; i += 256) { Cs[i] = 0.f; Ms[i] = 0.f; }
    __syncthreads();

    for (int t = 0; t < T_LEN; ++t) {
        // ---- (a) C_t = LN(f*C_prev + (1-f)*cand); wave w owns row b=w ----
        {
            const int b = w;
            size_t gbase = ((size_t)(bbase + b) * T_LEN + t) * 512;
            if (t + 1 < T_LEN) {   // prefetch next step's gate rows (2KB each)
                __builtin_prefetch((const char*)(fg + gbase + 512) + lane * 64, 0, 1);
                __builtin_prefetch((const char*)(cg + gbase + 512) + lane * 64, 0, 1);
                __builtin_prefetch((const char*)(ug + gbase + 512) + lane * 64, 0, 1);
            }
            float e[16];
            float s1 = 0.f, s2 = 0.f;
#pragma unroll
            for (int q = 0; q < 16; ++q) {
                int j = lane + q * 32;
                float f  = fg[gbase + j];
                float cd = cg[gbase + j];
                float vv = f * Cs[b * 512 + j] + (1.f - f) * cd;
                e[q] = vv; s1 += vv; s2 += vv * vv;
            }
#pragma unroll
            for (int off = 16; off; off >>= 1) {
                s1 += __shfl_xor(s1, off, 32);
                s2 += __shfl_xor(s2, off, 32);
            }
            float mean = s1 * (1.f / 512.f);
            float var  = s2 * (1.f / 512.f) - mean * mean;
            float rstd = rsqrtf(var + 1e-3f);
#pragma unroll
            for (int q = 0; q < 16; ++q) {
                int j = lane + q * 32;
                float cn = (e[q] - mean) * rstd * gamma_c[j] + beta_c[j];
                Cs[b * 512 + j] = cn;
                Cb[b * 512 + j] = f2bf(cn);
            }
        }
        __syncthreads();

        // ---- (b) m_pre = tanh(C_t @ Wm)*u + (1-u)*m_prev; wave w: 64 cols ----
        {
            v8f acc[4];
#pragma unroll
            for (int ni = 0; ni < 4; ++ni) acc[ni] = {};
#pragma unroll
            for (int kb = 0; kb < 16; ++kb) {
                const unsigned short* ap = &Cb[lr * 512 + kb * 32 + hs * 16];
                v16bf afrag = mk_frag(*(const uint4*)ap, *(const uint4*)(ap + 8));
#pragma unroll
                for (int ni = 0; ni < 4; ++ni) {
                    int n = w * 64 + ni * 16 + lr;
                    const unsigned short* bp = (kb < 8)
                        ? &WmL[n * 256 + kb * 32 + hs * 16]                 // LDS half
                        : &wmT[(size_t)n * 512 + kb * 32 + hs * 16];        // streamed half
                    v16bf bfrag = mk_frag(*(const uint4*)bp, *(const uint4*)(bp + 8));
                    acc[ni] = __builtin_amdgcn_wmma_f32_16x16x32_bf16(
                        false, afrag, false, bfrag, (short)0, acc[ni], false, false);
                }
            }
#pragma unroll
            for (int ni = 0; ni < 4; ++ni) {
                int n = w * 64 + ni * 16 + lr;
#pragma unroll
                for (int j = 0; j < 8; ++j) {
                    int bl = j + hs * 8;
                    if (bl < 8) {                      // rows 8..15 are padding
                        float g  = fast_tanh(acc[ni][j]);
                        float u  = ug[((size_t)(bbase + bl) * T_LEN + t) * 512 + n];
                        float mp = Ms[bl * 512 + n];
                        Ms[bl * 512 + n] = g * u + (1.f - u) * mp;
                    }
                }
            }
        }
        __syncthreads();

        // ---- (c) m_t = LN(m_pre); h = tanh(C_t * m_t); wave w owns row w ----
        {
            const int b = w;
            float e[16];
            float s1 = 0.f, s2 = 0.f;
#pragma unroll
            for (int q = 0; q < 16; ++q) {
                int j = lane + q * 32;
                float vv = Ms[b * 512 + j];
                e[q] = vv; s1 += vv; s2 += vv * vv;
            }
#pragma unroll
            for (int off = 16; off; off >>= 1) {
                s1 += __shfl_xor(s1, off, 32);
                s2 += __shfl_xor(s2, off, 32);
            }
            float mean = s1 * (1.f / 512.f);
            float var  = s2 * (1.f / 512.f) - mean * mean;
            float rstd = rsqrtf(var + 1e-3f);
            size_t obase = ((size_t)(bbase + b) * T_LEN + t) * 512;
#pragma unroll
            for (int q = 0; q < 16; ++q) {
                int j = lane + q * 32;
                float mn = (e[q] - mean) * rstd * gamma_m[j] + beta_m[j];
                Ms[b * 512 + j] = mn;
                out[obase + j] = fast_tanh(Cs[b * 512 + j] * mn);
            }
        }
        __syncthreads();
    }
}

// ---------------------------------------------------------------------------
extern "C" void kernel_launch(void* const* d_in, const int* in_sizes, int n_in,
                              void* d_out, int out_size, void* d_ws, size_t ws_size,
                              hipStream_t stream) {
    const float* x       = (const float*)d_in[0];
    const float* gk      = (const float*)d_in[1];
    const float* gb      = (const float*)d_in[2];
    const float* wm      = (const float*)d_in[3];
    const float* gamma_c = (const float*)d_in[4];
    const float* beta_c  = (const float*)d_in[5];
    const float* gamma_m = (const float*)d_in[6];
    const float* beta_m  = (const float*)d_in[7];
    float* out = (float*)d_out;

    char* ws = (char*)d_ws;
    unsigned short* xb  = (unsigned short*)ws; ws += (size_t)M_TOTAL * D_DIM * 2; // 64 MB
    unsigned short* wgT = (unsigned short*)ws; ws += (size_t)K3U * D_DIM * 2;     // 1.5 MB
    unsigned short* wmT = (unsigned short*)ws; ws += (size_t)U_DIM * U_DIM * 2;   // 0.5 MB
    float* fo = (float*)ws; ws += (size_t)M_TOTAL * U_DIM * 4;                    // 128 MB
    float* uo = (float*)ws; ws += (size_t)M_TOTAL * U_DIM * 4;                    // 128 MB
    float* co = (float*)ws;                                                       // 128 MB

    // Phase 0
    int n8 = M_TOTAL * D_DIM / 8;
    cvt_bf16_kernel<<<(n8 + 255) / 256, 256, 0, stream>>>(x, xb, n8);
    tconv_kernel<<<(D_DIM * K3U + 255) / 256, 256, 0, stream>>>(gk, wgT, D_DIM, K3U);
    tconv_kernel<<<(U_DIM * U_DIM + 255) / 256, 256, 0, stream>>>(wm, wmT, U_DIM, U_DIM);

    // Phase 1
    dim3 g1(K3U / 128, M_TOTAL / 128);
    gates_gemm_kernel<<<g1, 256, 0, stream>>>(xb, wgT, gb, fo, uo, co);

    // Phase 2: 8 independent scan workgroups (8 batch rows each), 304KB LDS
    const int scan_lds = 512 * 256 * 2 + 2 * 8 * 512 * 4 + 16 * 512 * 2;  // 311296
    (void)hipFuncSetAttribute((const void*)sru_scan_kernel,
                              hipFuncAttributeMaxDynamicSharedMemorySize, 320 * 1024);
    sru_scan_kernel<<<8, 256, scan_lds, stream>>>(
        fo, uo, co, wmT, gamma_c, beta_c, gamma_m, beta_m, out);
}
